// WindowedAttention_472446403162
// MI455X (gfx1250) — matrix operand order repaired and checked by reading, not compiled, not run on hardware
//
#include <hip/hip_runtime.h>
#include <hip/hip_bf16.h>

typedef __attribute__((ext_vector_type(16))) _Float16 v16h;
typedef __attribute__((ext_vector_type(8)))  _Float16 v8h;
typedef __attribute__((ext_vector_type(4)))  _Float16 v4h;
typedef __attribute__((ext_vector_type(8)))  float    v8f;

#define WMMA16(a, b, c) __builtin_amdgcn_wmma_f32_16x16x32_f16( \
    false, (a), false, (b), (short)0, (c), false, false)

// ---------------- problem constants ----------------
#define BATCH   8
#define SEQ     3136      // 56*56
#define CDIM    768
#define NHEAD   12
#define HDIM    64
#define WSZ     14
#define NWIN    16        // 4*4 windows
#define NTOK    196       // 14*14
#define MROWS   25088     // BATCH*SEQ
#define CQKV    2304
#define KSTEPS  (CDIM / 32)   // 24

// workspace layout (f16): qkv [3][B][16][12][196][64], then attn_out [MROWS][CDIM]
#define QKV_ELEMS   ((size_t)3 * BATCH * NWIN * NHEAD * NTOK * HDIM)   // 57,802,752
#define ATTN_OFF_B  (QKV_ELEMS * 2)                                    // bytes

// ---------------- CDNA5 helpers ----------------
// raw LDS byte offset (addrspace(3) ptrtoint) for async-to-LDS vdst operand
__device__ __forceinline__ uint32_t lds_addr(const void* p) {
  return (uint32_t)(uintptr_t)(__attribute__((address_space(3))) const void*)p;
}

// GLOBAL_LOAD_ASYNC_TO_LDS_B128, GV mode: per-lane 16B global -> LDS copy
__device__ __forceinline__ void async_copy_b128(uint32_t lds_byte_off,
                                                const void* gaddr) {
  unsigned long long ga = (unsigned long long)(uintptr_t)gaddr;
  asm volatile("global_load_async_to_lds_b128 %0, %1, off"
               :: "v"(lds_byte_off), "v"(ga) : "memory");
}
__device__ __forceinline__ void wait_asynccnt0() {
  asm volatile("s_wait_asynccnt 0x0" ::: "memory");
}

// ---------------- fragment helpers (wave32, 16x16x32 f16) ----------------
// A 16x32: lane<16 -> M=m0+lane, K={k0..k0+7, k0+16..k0+23}; lane>=16 -> K+8
__device__ __forceinline__ v16h frag_a16(const _Float16* p, int m0, int mmax,
                                         int stride, int k0) {
  int lane = threadIdx.x & 31;
  int m = m0 + (lane & 15);
  if (m > mmax) m = mmax;
  int ks = (lane >> 4) << 3;   // 0 or 8
  const _Float16* r = p + (size_t)m * stride + k0 + ks;
  v8h lo = *(const v8h*)(r);
  v8h hi = *(const v8h*)(r + 16);
  v16h f;
#pragma unroll
  for (int i = 0; i < 8; ++i) { f[i] = lo[i]; f[i + 8] = hi[i]; }
  return f;
}

// B 32x16 from N-major buffer: lane<16 -> N=n0+lane, K={k0..k0+7, k0+16..k0+23}; lane>=16 -> K+8
__device__ __forceinline__ v16h frag_bT(const _Float16* p, int n0, int nmax,
                                        int stride, int k0) {
  int lane = threadIdx.x & 31;
  int n = n0 + (lane & 15);
  if (n > nmax) n = nmax;
  int ks = (lane >> 4) << 3;   // 0 or 8
  const _Float16* r = p + (size_t)n * stride + k0 + ks;
  v8h lo = *(const v8h*)(r);
  v8h hi = *(const v8h*)(r + 16);
  v16h f;
#pragma unroll
  for (int i = 0; i < 8; ++i) { f[i] = lo[i]; f[i + 8] = hi[i]; }
  return f;
}

// ---------------- kernel 1: qkv = x @ w_qkv, scatter to windowed f16 layout ----
__device__ __forceinline__ void stage_k1(const float* __restrict__ x,
                                         const float* __restrict__ w,
                                         _Float16* As, _Float16* Bt,
                                         int m0, int n0, int k0, int tid) {
  {
    int r = tid >> 3, c = (tid & 7) * 4;
#pragma unroll
    for (int rr = 0; rr < 4; ++rr) {
      int row = r + rr * 32;
      float4 v = *(const float4*)(x + (size_t)(m0 + row) * CDIM + k0 + c);
      _Float16* d = As + row * 32 + c;
      d[0] = (_Float16)v.x; d[1] = (_Float16)v.y;
      d[2] = (_Float16)v.z; d[3] = (_Float16)v.w;
    }
  }
  {
    int r = tid >> 5, c = (tid & 31) * 4;
#pragma unroll
    for (int rr = 0; rr < 4; ++rr) {
      int kk = r + rr * 8;
      float4 v = *(const float4*)(w + (size_t)(k0 + kk) * CQKV + n0 + c);
      Bt[(c + 0) * 32 + kk] = (_Float16)v.x;
      Bt[(c + 1) * 32 + kk] = (_Float16)v.y;
      Bt[(c + 2) * 32 + kk] = (_Float16)v.z;
      Bt[(c + 3) * 32 + kk] = (_Float16)v.w;
    }
  }
  // prefetch two k-steps ahead (global_prefetch_b8)
  if (k0 + 32 < CDIM) {
    __builtin_prefetch(x + (size_t)(m0 + (tid >> 1)) * CDIM + k0 + 32, 0, 0);
    __builtin_prefetch(w + (size_t)(k0 + 32 + (tid & 31)) * CQKV + n0 + (tid >> 6) * 32, 0, 0);
  }
}

__global__ __launch_bounds__(256) void qkv_gemm_kernel(
    const float* __restrict__ x, const float* __restrict__ w,
    _Float16* __restrict__ qkv) {
  __shared__ _Float16 As[2][128 * 32];
  __shared__ _Float16 Bt[2][128 * 32];

  int tid  = threadIdx.x;
  int wave = tid >> 5;
  int lane = tid & 31;
  int m0 = blockIdx.x * 128;
  int n0 = blockIdx.y * 128;
  int wm = (wave & 3) * 32;
  int wn = (wave >> 2) * 64;

  v8f acc[2][4];
#pragma unroll
  for (int i = 0; i < 2; ++i)
#pragma unroll
    for (int j = 0; j < 4; ++j) { v8f z = {}; acc[i][j] = z; }

  stage_k1(x, w, As[0], Bt[0], m0, n0, 0, tid);
#pragma unroll 1
  for (int it = 0; it < KSTEPS; ++it) {
    __syncthreads();
    int cur = it & 1;
    if (it + 1 < KSTEPS)
      stage_k1(x, w, As[1 - cur], Bt[1 - cur], m0, n0, (it + 1) * 32, tid);
    v16h a[2], b[4];
#pragma unroll
    for (int i = 0; i < 2; ++i) a[i] = frag_a16(As[cur], wm + i * 16, 127, 32, 0);
#pragma unroll
    for (int j = 0; j < 4; ++j) b[j] = frag_bT(Bt[cur], wn + j * 16, 127, 32, 0);
#pragma unroll
    for (int i = 0; i < 2; ++i)
#pragma unroll
      for (int j = 0; j < 4; ++j) acc[i][j] = WMMA16(a[i], b[j], acc[i][j]);
  }

  // scatter epilogue into [which][b][win][head][tok][d] f16
  int nb = lane & 15;
  int mh = (lane >> 4) * 8;
#pragma unroll
  for (int i = 0; i < 2; ++i)
#pragma unroll
    for (int j = 0; j < 4; ++j)
#pragma unroll
      for (int e = 0; e < 8; ++e) {
        int m = m0 + wm + i * 16 + mh + e;
        int c = n0 + wn + j * 16 + nb;
        int which = c / CDIM;      int rem = c - which * CDIM;
        int head  = rem >> 6;      int d   = rem & 63;
        int bb  = m / SEQ;         int pos = m - bb * SEQ;
        int hh  = pos / 56;        int ww  = pos - hh * 56;
        int hn  = hh / WSZ;        int ii  = hh - hn * WSZ;
        int wnn = ww / WSZ;        int jj  = ww - wnn * WSZ;
        int lwin = hn * 4 + wnn;
        size_t idx = ((size_t)(((which * BATCH + bb) * NWIN + lwin) * NHEAD + head))
                       * (NTOK * HDIM) + (ii * WSZ + jj) * HDIM + d;
        qkv[idx] = (_Float16)acc[i][j][e];
      }
}

// ---------------- kernel 2: per-(b,window,head) attention ----------------
// LDS: S f32 [196][208] | P f16 [208][224] | vT f16 [64][224]  = 284,928 B
#define S_BYTES  (196 * 208 * 4)
#define P_BYTES  (208 * 224 * 2)
#define VT_BYTES (64 * 224 * 2)
#define ATTN_SMEM (S_BYTES + P_BYTES + VT_BYTES)

__global__ __launch_bounds__(256) void win_attn_kernel(
    const _Float16* __restrict__ qkv, _Float16* __restrict__ attn_out) {
  extern __shared__ char smem[];
  float*    S  = (float*)smem;
  _Float16* P  = (_Float16*)(smem + S_BYTES);
  _Float16* vT = (_Float16*)(smem + S_BYTES + P_BYTES);

  int wg = blockIdx.x;
  int h  = wg % NHEAD;
  int l  = (wg / NHEAD) & (NWIN - 1);
  int b  = wg / (NHEAD * NWIN);
  size_t base = (size_t)((b * NWIN + l) * NHEAD + h) * (NTOK * HDIM);
  size_t one  = (size_t)BATCH * NWIN * NHEAD * NTOK * HDIM;
  const _Float16* q = qkv + base;
  const _Float16* k = qkv + one + base;
  const _Float16* v = qkv + 2 * one + base;

  int tid = threadIdx.x;
  // stage V transposed [64][224], zero-pad keys 196..223
  for (int idx = tid; idx < 64 * 224; idx += 256) {
    int d = idx / 224, m = idx - d * 224;
    vT[idx] = (m < NTOK) ? v[(size_t)m * HDIM + d] : (_Float16)0.f;
  }
  __syncthreads();

  int wave = tid >> 5, lane = tid & 31;
  int nb = lane & 15, mh = (lane >> 4) * 8;

  // pass 1: S = (q k^T) * scale, 13x13 tiles of 16x16, K = 64 (2 wmma)
  for (int wrk = wave; wrk < 169; wrk += 8) {
    int s = wrk / 13, t = wrk - s * 13;
    v16h a0 = frag_a16(q, s * 16, NTOK - 1, HDIM, 0);
    v16h a1 = frag_a16(q, s * 16, NTOK - 1, HDIM, 32);
    v16h b0 = frag_bT(k, t * 16, NTOK - 1, HDIM, 0);
    v16h b1 = frag_bT(k, t * 16, NTOK - 1, HDIM, 32);
    v8f c = {};
    c = WMMA16(a0, b0, c);
    c = WMMA16(a1, b1, c);
#pragma unroll
    for (int e = 0; e < 8; ++e) {
      int m = s * 16 + mh + e;
      if (m < NTOK) S[m * 208 + t * 16 + nb] = c[e] * 0.125f;  // hd^-0.5
    }
  }
  __syncthreads();

  // softmax: one thread per row, write P f16 with K-pad zeros
  if (tid < NTOK) {
    float* row = S + tid * 208;
    float mx = -1e30f;
    for (int c2 = 0; c2 < NTOK; ++c2) mx = fmaxf(mx, row[c2]);
    float sum = 0.f;
    for (int c2 = 0; c2 < NTOK; ++c2) sum += __expf(row[c2] - mx);
    float inv = 1.f / sum;
    _Float16* pr = P + tid * 224;
    for (int c2 = 0; c2 < NTOK; ++c2) pr[c2] = (_Float16)(__expf(row[c2] - mx) * inv);
    for (int c2 = NTOK; c2 < 224; ++c2) pr[c2] = (_Float16)0.f;
  } else if (tid < 208) {
    _Float16* pr = P + tid * 224;
    for (int c2 = 0; c2 < 224; ++c2) pr[c2] = (_Float16)0.f;
  }
  __syncthreads();

  // pass 2: out = P @ V, 13x4 output tiles, K = 224 (7 wmma each)
  int hn = l >> 2, wnn = l & 3;
  for (int wrk = wave; wrk < 52; wrk += 8) {
    int s = wrk >> 2, t = wrk & 3;
    v8f c = {};
#pragma unroll
    for (int kk = 0; kk < 7; ++kk) {
      v16h a  = frag_a16(P, s * 16, 207, 224, kk * 32);
      v16h bb = frag_bT(vT, t * 16, 63, 224, kk * 32);
      c = WMMA16(a, bb, c);
    }
#pragma unroll
    for (int e = 0; e < 8; ++e) {
      int m = s * 16 + mh + e;
      if (m < NTOK) {
        int ii = m / WSZ, jj = m - ii * WSZ;
        int grow = b * SEQ + (hn * WSZ + ii) * 56 + wnn * WSZ + jj;
        int gcol = h * HDIM + t * 16 + nb;
        attn_out[(size_t)grow * CDIM + gcol] = (_Float16)c[e];
      }
    }
  }
}

// ---------------- kernel 3: out = attn @ w_proj + b_proj (f32 out) ----------
// A tile is f16 already -> stage via GLOBAL_LOAD_ASYNC_TO_LDS_B128 (ASYNCcnt).
__device__ __forceinline__ void stage_k3(const _Float16* __restrict__ a,
                                         const float* __restrict__ w,
                                         _Float16* As, _Float16* Bt,
                                         int m0, int n0, int k0, int tid) {
  // A: 128x32 f16 = 8KB = 512 x 16B chunks; 2 per thread, async copy
#pragma unroll
  for (int it = 0; it < 2; ++it) {
    int qq  = tid + it * 256;            // 0..511
    int row = qq >> 2;
    int part = (qq & 3) * 8;             // halves
    const _Float16* g = a + (size_t)(m0 + row) * CDIM + k0 + part;
    async_copy_b128(lds_addr(As + row * 32 + part), g);
  }
  // B: f32 -> f16 transposed staging (regular path)
  int r = tid >> 5, c = (tid & 31) * 4;
#pragma unroll
  for (int rr = 0; rr < 4; ++rr) {
    int kk = r + rr * 8;
    float4 v = *(const float4*)(w + (size_t)(k0 + kk) * CDIM + n0 + c);
    Bt[(c + 0) * 32 + kk] = (_Float16)v.x;
    Bt[(c + 1) * 32 + kk] = (_Float16)v.y;
    Bt[(c + 2) * 32 + kk] = (_Float16)v.z;
    Bt[(c + 3) * 32 + kk] = (_Float16)v.w;
  }
  wait_asynccnt0();   // async tile resident in LDS before the next barrier
}

__global__ __launch_bounds__(256) void proj_gemm_kernel(
    const _Float16* __restrict__ a, const float* __restrict__ w,
    const float* __restrict__ bias, float* __restrict__ out) {
  __shared__ _Float16 As[2][128 * 32];
  __shared__ _Float16 Bt[2][128 * 32];

  int tid  = threadIdx.x;
  int wave = tid >> 5;
  int lane = tid & 31;
  int m0 = blockIdx.x * 128;
  int n0 = blockIdx.y * 128;
  int wm = (wave & 3) * 32;
  int wn = (wave >> 2) * 64;

  v8f acc[2][4];
#pragma unroll
  for (int i = 0; i < 2; ++i)
#pragma unroll
    for (int j = 0; j < 4; ++j) { v8f z = {}; acc[i][j] = z; }

  stage_k3(a, w, As[0], Bt[0], m0, n0, 0, tid);
#pragma unroll 1
  for (int it = 0; it < KSTEPS; ++it) {
    __syncthreads();
    int cur = it & 1;
    if (it + 1 < KSTEPS)
      stage_k3(a, w, As[1 - cur], Bt[1 - cur], m0, n0, (it + 1) * 32, tid);
    v16h af[2], bf[4];
#pragma unroll
    for (int i = 0; i < 2; ++i) af[i] = frag_a16(As[cur], wm + i * 16, 127, 32, 0);
#pragma unroll
    for (int j = 0; j < 4; ++j) bf[j] = frag_bT(Bt[cur], wn + j * 16, 127, 32, 0);
#pragma unroll
    for (int i = 0; i < 2; ++i)
#pragma unroll
      for (int j = 0; j < 4; ++j) acc[i][j] = WMMA16(af[i], bf[j], acc[i][j]);
  }

  int nb = lane & 15;
  int mh = (lane >> 4) * 8;
#pragma unroll
  for (int i = 0; i < 2; ++i)
#pragma unroll
    for (int j = 0; j < 4; ++j)
#pragma unroll
      for (int e = 0; e < 8; ++e) {
        int m = m0 + wm + i * 16 + mh + e;
        int c = n0 + wn + j * 16 + nb;
        out[(size_t)m * CDIM + c] = acc[i][j][e] + bias[c];
      }
}

// ---------------- launcher ----------------
extern "C" void kernel_launch(void* const* d_in, const int* in_sizes, int n_in,
                              void* d_out, int out_size, void* d_ws, size_t ws_size,
                              hipStream_t stream) {
  (void)in_sizes; (void)n_in; (void)out_size; (void)ws_size;
  const float* x      = (const float*)d_in[0];
  const float* w_qkv  = (const float*)d_in[1];
  const float* w_proj = (const float*)d_in[2];
  const float* b_proj = (const float*)d_in[3];

  _Float16* qkv  = (_Float16*)d_ws;
  _Float16* attn = (_Float16*)((char*)d_ws + ATTN_OFF_B);
  float*    out  = (float*)d_out;

  // allow >64KB dynamic LDS for the attention kernel (deterministic, capture-safe)
  (void)hipFuncSetAttribute((const void*)win_attn_kernel,
                            hipFuncAttributeMaxDynamicSharedMemorySize, ATTN_SMEM);

  dim3 g1(MROWS / 128, CQKV / 128);
  qkv_gemm_kernel<<<g1, 256, 0, stream>>>(x, w_qkv, qkv);

  win_attn_kernel<<<BATCH * NWIN * NHEAD, 256, ATTN_SMEM, stream>>>(qkv, attn);

  dim3 g3(MROWS / 128, CDIM / 128);
  proj_gemm_kernel<<<g3, 256, 0, stream>>>(attn, w_proj, b_proj, out);
}
